// GlobalBaseModule_61976378081890
// MI455X (gfx1250) — matrix operand order repaired and checked by reading, not compiled, not run on hardware
//
#include <hip/hip_runtime.h>
#include <stdint.h>

typedef __attribute__((ext_vector_type(16))) __bf16 v16bf;
typedef __attribute__((ext_vector_type(8)))  float  v8f;

#define DX    64
#define DHID  128
#define DOUT  256
#define NSEG  16
#define W1_FRAGS 24            /* 8 coltiles * 3 k-chunks */
#define W2_FRAGS 64            /* 16 coltiles * 4 k-chunks */
#define W1_DW (W1_FRAGS * 256) /* dwords: 32 lanes * 8 dwords per frag */
#define W2_DW (W2_FRAGS * 256)

__device__ __forceinline__ unsigned int f2bfbits_rne(float f) {
    unsigned int u = __float_as_uint(f);
    u += 0x7FFFu + ((u >> 16) & 1u);
    return u >> 16;
}
__device__ __forceinline__ unsigned int pack2_rne(float lo, float hi) {
    return f2bfbits_rne(lo) | (f2bfbits_rne(hi) << 16);
}
// Truncating pack: one v_perm_b32 -> {hi[31:16], lo[31:16]}
__device__ __forceinline__ unsigned int pack2t(float lo, float hi) {
    return __builtin_amdgcn_perm(__float_as_uint(hi), __float_as_uint(lo), 0x07060302u);
}

union BFrag {
    v16bf v;
    uint4 q[2];
    unsigned int u[8];
};

__global__ void init_out(float* out, int out_size) {
    int i = blockIdx.x * 256 + threadIdx.x;
    const int tail = NSEG * DOUT + NSEG * 3;   // pooled + pos_out
    if (i < tail) {
        out[i] = 0.0f;                         // ReLU>=0, so 0 is the identity for the max-pool
    } else if (i < out_size) {
        out[i] = (float)(i - tail);            // batch_out = arange(16)
    }
}

// Repack W1/W2 (f32 row-major) into bf16 WMMA B-fragments (32x16; lanes 0-15: K=0..15,
// lanes 16-31: K=16..31; 2 packed bf16 per dword). Fragment f = 256 dwords: lane*8 + v.
// One-time cost -> keep RNE rounding here.
__global__ void pack_weights(const float* __restrict__ W1, const float* __restrict__ W2,
                             unsigned int* __restrict__ ws) {
    int t = blockIdx.x * 256 + threadIdx.x;
    if (t < W1_DW) {
        int v = t & 7, lane = (t >> 3) & 31, f = t >> 8;
        int n = f / 3, c = f % 3;
        int col = n * 16 + (lane & 15);
        int kb  = 32 * c + ((lane < 16) ? 0 : 16);
        int k0 = kb + 2 * v, k1 = k0 + 1;
        float f0 = (k0 < DX + 3) ? W1[k0 * DHID + col] : 0.0f;
        float f1 = (k1 < DX + 3) ? W1[k1 * DHID + col] : 0.0f;
        ws[t] = pack2_rne(f0, f1);
    } else if (t < W1_DW + W2_DW) {
        int s = t - W1_DW;
        int v = s & 7, lane = (s >> 3) & 31, f = s >> 8;
        int n = f / 4, c = f % 4;
        int col = n * 16 + (lane & 15);
        int kb  = 32 * c + ((lane < 16) ? 0 : 16);
        int k0 = kb + 2 * v;
        ws[t] = pack2_rne(W2[k0 * DOUT + col], W2[(k0 + 1) * DOUT + col]);
    }
}

__global__ void __launch_bounds__(256) pointnet_wmma(
    const float* __restrict__ x, const float* __restrict__ pos,
    const long long* __restrict__ batch,
    const float* __restrict__ b1, const float* __restrict__ b2,
    const unsigned int* __restrict__ ws, float* __restrict__ out,
    int numTiles)
{
    __shared__ unsigned short hbuf[8][16 * DHID];   // per-wave 16x128 bf16 bounce tile (32 KB)

    const int lane = threadIdx.x & 31;
    const int wave = threadIdx.x >> 5;
    const int hi   = lane >> 4;        // 0: lanes 0-15, 1: lanes 16-31
    const int lcol = lane & 15;
    const int kb2  = hi ? 16 : 0;      // A-matrix K sub-offset per lane half

    unsigned short* hrow = hbuf[wave];
    const int wavesTotal = gridDim.x * 8;

    for (int tile = blockIdx.x * 8 + wave; tile < numTiles; tile += wavesTotal) {
        const int row0 = tile * 16;
        const int r    = row0 + lcol;       // this lane's A-matrix row

        // ---- A1 fragments: 16x32 bf16, K=0..95 (x[0..63], pos[64..66], zero pad) ----
        BFrag a1[3];
#pragma unroll
        for (int c = 0; c < 2; c++) {
            const float* px = x + (size_t)r * DX + c * 32 + kb2;
            float4 f0 = *(const float4*)(px);
            float4 f1 = *(const float4*)(px + 4);
            float4 f2 = *(const float4*)(px + 8);
            float4 f3 = *(const float4*)(px + 12);
            a1[c].u[0] = pack2t(f0.x, f0.y); a1[c].u[1] = pack2t(f0.z, f0.w);
            a1[c].u[2] = pack2t(f1.x, f1.y); a1[c].u[3] = pack2t(f1.z, f1.w);
            a1[c].u[4] = pack2t(f2.x, f2.y); a1[c].u[5] = pack2t(f2.z, f2.w);
            a1[c].u[6] = pack2t(f3.x, f3.y); a1[c].u[7] = pack2t(f3.z, f3.w);
        }
        {
            float p0 = 0.f, p1 = 0.f, p2 = 0.f;
            if (!hi) { const float* pp = pos + (size_t)r * 3; p0 = pp[0]; p1 = pp[1]; p2 = pp[2]; }
            a1[2].u[0] = pack2t(p0, p1); a1[2].u[1] = pack2t(p2, 0.f);
            a1[2].u[2] = 0; a1[2].u[3] = 0; a1[2].u[4] = 0;
            a1[2].u[5] = 0; a1[2].u[6] = 0; a1[2].u[7] = 0;
        }

        // ---- segment ids for the 16 rows of this tile (batch is sorted) ----
        int myseg = 0;
        if (!hi) myseg = (int)batch[row0 + lcol];
        const int  segFirst = __shfl(myseg, 0, 32);
        const int  segLast  = __shfl(myseg, 15, 32);
        const bool uniformSeg = (segFirst == segLast);
        int segRow[8];
#pragma unroll
        for (int v = 0; v < 8; v++) segRow[v] = __shfl(myseg, v + (hi ? 8 : 0), 32);

        // ---- layer 1: [16x96] x [96x128], bias+ReLU, bf16 -> LDS (C->A layout transpose) ----
        // unroll 1: keep only one coltile's B fragments (24 VGPRs) live -> no spills
#pragma unroll 1
        for (int n = 0; n < 8; n++) {
            BFrag b[3];
            const uint4* bp = (const uint4*)(ws + (n * 3 * 256 + lane * 8));
#pragma unroll
            for (int c = 0; c < 3; c++) {             // contiguous: one load clause
                b[c].q[0] = bp[c * 64];               // 256 dwords = 64 uint4 per frag
                b[c].q[1] = bp[c * 64 + 1];
            }
            const float bias = b1[n * 16 + lcol];
            v8f acc = {0.f, 0.f, 0.f, 0.f, 0.f, 0.f, 0.f, 0.f};
#pragma unroll
            for (int c = 0; c < 3; c++)
                acc = __builtin_amdgcn_wmma_f32_16x16x32_bf16(false, a1[c].v, false, b[c].v,
                                                              (short)0, acc, false, false);
            const int col = n * 16 + lcol;
            unsigned short* hp = hrow + (hi ? 8 : 0) * DHID + col;
#pragma unroll
            for (int v = 0; v < 8; v++) {
                float h = fmaxf(acc[v] + bias, 0.0f);
                // truncating bf16: store high 16 bits (compiler emits ds_store_b16_d16_hi)
                hp[v * DHID] = (unsigned short)(__float_as_uint(h) >> 16);
            }
        }

        // ---- A2 fragments from LDS: 16x32 bf16, 4 chunks over K=0..127 ----
        BFrag a2[4];
#pragma unroll
        for (int c = 0; c < 4; c++) {
            a2[c].q[0] = *(const uint4*)&hrow[lcol * DHID + c * 32 + kb2];
            a2[c].q[1] = *(const uint4*)&hrow[lcol * DHID + c * 32 + 8 + kb2];
        }

        // ---- layer 2: [16x128] x [128x256], bias+ReLU, segment max-pool ----
#pragma unroll 1
        for (int n = 0; n < 16; n++) {
            BFrag b[4];
            const uint4* bp = (const uint4*)(ws + (W1_DW + n * 4 * 256 + lane * 8));
#pragma unroll
            for (int c = 0; c < 4; c++) {
                b[c].q[0] = bp[c * 64];
                b[c].q[1] = bp[c * 64 + 1];
            }
            const float bias = b2[n * 16 + lcol];
            v8f acc = {0.f, 0.f, 0.f, 0.f, 0.f, 0.f, 0.f, 0.f};
#pragma unroll
            for (int c = 0; c < 4; c++)
                acc = __builtin_amdgcn_wmma_f32_16x16x32_bf16(false, a2[c].v, false, b[c].v,
                                                              (short)0, acc, false, false);
            const int col = n * 16 + lcol;
            if (uniformSeg) {
                // reduce 16 rows in-wave; ReLU>=0 so u32 bit-pattern max == float max
                float m = fmaxf(acc[0] + bias, 0.0f);
#pragma unroll
                for (int v = 1; v < 8; v++) m = fmaxf(m, fmaxf(acc[v] + bias, 0.0f));
                m = fmaxf(m, __shfl_xor(m, 16, 32));
                if (!hi)
                    atomicMax((unsigned int*)&out[segFirst * DOUT + col], __float_as_uint(m));
            } else {
#pragma unroll
                for (int v = 0; v < 8; v++) {
                    float h = fmaxf(acc[v] + bias, 0.0f);
                    atomicMax((unsigned int*)&out[segRow[v] * DOUT + col], __float_as_uint(h));
                }
            }
        }
    }
}

extern "C" void kernel_launch(void* const* d_in, const int* in_sizes, int n_in,
                              void* d_out, int out_size, void* d_ws, size_t ws_size,
                              hipStream_t stream) {
    const float*     x     = (const float*)d_in[0];
    const float*     pos   = (const float*)d_in[1];
    const long long* batch = (const long long*)d_in[2];
    const float*     W1    = (const float*)d_in[3];
    const float*     b1    = (const float*)d_in[4];
    const float*     W2    = (const float*)d_in[5];
    const float*     b2    = (const float*)d_in[6];
    float*           out   = (float*)d_out;
    unsigned int*    ws    = (unsigned int*)d_ws;

    const int N        = in_sizes[0] / DX;
    const int numTiles = N / 16;

    init_out<<<(out_size + 255) / 256, 256, 0, stream>>>(out, out_size);
    pack_weights<<<(W1_DW + W2_DW + 255) / 256, 256, 0, stream>>>(W1, W2, ws);
    pointnet_wmma<<<2048, 256, 0, stream>>>(x, pos, batch, b1, b2, ws, out, numTiles);
}
